// SMN_70935679861140
// MI455X (gfx1250) — matrix operand order repaired and checked by reading, not compiled, not run on hardware
//
#include <hip/hip_runtime.h>

// B1=16, B2=16, N=32, D=64, H=2. One workgroup per (b1,b2) set; 256 threads = 8 waves (wave32).

typedef __attribute__((ext_vector_type(16))) _Float16 v16h;
typedef __attribute__((ext_vector_type(8)))  float    v8f;

#define LD64  65
#define LD128 132   // multiple of 4 -> 16B-aligned rows for float4 LDS loads; 132%64=4 keeps banks spread
#define LD32  33

// Branch-free tanh-gelu: gelu(v) = v * sigmoid(2c*(v + 0.044715 v^3)), c = sqrt(2/pi).
// exp(-y) = exp2(z * -2c*log2(e)); uses v_exp_f32 + v_rcp_f32 (no OCML branches).
__device__ __forceinline__ float gelu_f(float v) {
    float z = v * (1.0f + 0.044715f * v * v);             // v + 0.044715 v^3
    float e = __builtin_amdgcn_exp2f(z * -2.302208f);     // -2*sqrt(2/pi)*log2(e)
    return v * __builtin_amdgcn_rcpf(1.0f + e);
}

// A fragment (16x32 f16) from f32 row-major [M x ld]; ISA 7.12.2 16-bit A layout:
// lane: M = m0 + lane%16 ; half h -> K = (lane>=16 ? 8:0) + h + (h>=8 ? 8:0)
__device__ __forceinline__ v16h frag_a_f32(const float* S, int ld, int m0, int k0) {
    const int lane = threadIdx.x & 31;
    const int m  = m0 + (lane & 15);
    const int kb = (lane & 16) ? 8 : 0;
    v16h a;
#pragma unroll
    for (int h = 0; h < 16; ++h) {
        int k = k0 + kb + h + ((h & 8) ? 8 : 0);
        a[h] = (_Float16)S[m * ld + k];
    }
    return a;
}

// B fragment (32x16 f16) from f32 row-major [K x ld]; same K layout as A:
// half h -> K = (lane>=16 ? 8:0) + h + (h>=8 ? 8:0); N = n0 + lane%16.
__device__ __forceinline__ v16h frag_b_f32(const float* W, int ldw, int k0, int n0) {
    const int lane = threadIdx.x & 31;
    const int n  = n0 + (lane & 15);
    const int kb = (lane & 16) ? 8 : 0;
    v16h b;
#pragma unroll
    for (int h = 0; h < 16; ++h) {
        int k = k0 + kb + h + ((h & 8) ? 8 : 0);
        b[h] = (_Float16)W[k * ldw + n];
    }
    return b;
}

// D store: lane col N = n0 + lane%16 ; VGPR r -> row m0 + r + (lane>=16 ? 8 : 0)
__device__ __forceinline__ void store_d(float* S, int ld, int m0, int n0, v8f d) {
    const int lane = threadIdx.x & 31;
    const int n  = n0 + (lane & 15);
    const int mo = (lane & 16) ? 8 : 0;
#pragma unroll
    for (int r = 0; r < 8; ++r)
        S[(m0 + r + mo) * ld + n] = d[r];
}

// Fully-unrolled tiled wave GEMM: Dst[MxN] = A[MxK] * B[KxN] (raw). 8 waves split tiles.
template <int M, int N, int K>
__device__ __forceinline__ void mm_wave(const float* A, int lda,
                                        const float* B, int ldb,
                                        float* Dst, int ldd, int wave) {
    constexpr int NT = N >> 4;
    constexpr int KT = K >> 5;
    constexpr int T  = (M >> 4) * NT;   // 8 or 16 in this kernel
    constexpr int RT = T >> 3;          // tiles per wave
#pragma unroll
    for (int r = 0; r < RT; ++r) {
        const int t  = wave + (r << 3);
        const int m0 = (t / NT) << 4;
        const int n0 = (t % NT) << 4;
        v8f acc = {};
#pragma unroll
        for (int kk = 0; kk < KT; ++kk) {
            v16h a = frag_a_f32(A, lda, m0, kk << 5);
            v16h b = frag_b_f32(B, ldb, kk << 5, n0);
            acc = __builtin_amdgcn_wmma_f32_16x16x32_f16(
                false, a, false, b, (short)0, acc, false, false);
        }
        store_d(Dst, ldd, m0, n0, acc);
    }
}

__global__ __launch_bounds__(256) void smn_fused(
    const float* __restrict__ x,   const float* __restrict__ x_size,
    const float* __restrict__ w1a, const float* __restrict__ b1a,
    const float* __restrict__ w1b, const float* __restrict__ b1b,
    const float* __restrict__ w2a, const float* __restrict__ b2a,
    const float* __restrict__ w2b, const float* __restrict__ b2b,
    const float* __restrict__ w3,  const float* __restrict__ b3,
    float* __restrict__ out)
{
    __shared__ float s_x [32 * LD64];                 // original x (kept for residual)
    __shared__ float s_xn[32 * LD64];                 // layernormed x; reused for raw proj
    __shared__ float s_m1[32 * LD64];                 // x_mlp1
    __shared__ __align__(16) float s_PH[32 * LD128];  // h1, then P = x_mlp1@w2a + b2a
    __shared__ __align__(16) float s_Q [32 * LD128];  // Q = xn@w2a
    __shared__ float s_tT[32 * LD32];                 // t transposed: tT[j][i]
    __shared__ __align__(16) float s_red[256];        // reductions; later u[k] (k<128)
    __shared__ float s_rmask[32];                     // per-item mask

    const int tid  = threadIdx.x;
    const int wave = tid >> 5;
    const int set  = blockIdx.x;                 // b1*16 + b2
    const float* xs = x + (size_t)set * 2048;
    const float denom = x_size[set >> 4] * 64.0f;

    // ---- load x, partial sum for mean ----
    float psum = 0.f;
    for (int e = tid; e < 2048; e += 256) {
        float v = xs[e];
        s_x[(e >> 6) * LD64 + (e & 63)] = v;
        psum += v;
    }
    __syncthreads();

    // ---- per-item mask (row all-zero -> 0) ----
    if (tid < 32) {
        float any = 0.f;
        for (int d = 0; d < 64; ++d)
            if (s_x[tid * LD64 + d] != 0.f) any = 1.f;
        s_rmask[tid] = any;
    }

    // ---- mean over N*D ----
    s_red[tid] = psum;
    __syncthreads();
    for (int s = 128; s > 0; s >>= 1) {
        if (tid < s) s_red[tid] += s_red[tid + s];
        __syncthreads();
    }
    const float mean = s_red[0] / denom;
    __syncthreads();

    // ---- masked variance ----
    float vsum = 0.f;
    for (int e = tid; e < 2048; e += 256) {
        int i = e >> 6, d = e & 63;
        float df = s_x[i * LD64 + d] - mean;
        vsum += df * df * s_rmask[i];
    }
    s_red[tid] = vsum;
    __syncthreads();
    for (int s = 128; s > 0; s >>= 1) {
        if (tid < s) s_red[tid] += s_red[tid + s];
        __syncthreads();
    }
    const float inv = 1.0f / (sqrtf(s_red[0] / denom) + 1e-8f);
    __syncthreads();

    for (int e = tid; e < 2048; e += 256) {
        int i = e >> 6, d = e & 63;
        s_xn[i * LD64 + d] = (s_x[i * LD64 + d] - mean) * inv * s_rmask[i];
    }
    __syncthreads();

    // ---- MLP1a: xn(32x64) @ w1a(64x128) -> s_PH raw ----
    mm_wave<32, 128, 64>(s_xn, LD64, w1a, 128, s_PH, LD128, wave);
    __syncthreads();
    for (int e = tid; e < 4096; e += 256) {
        int i = e >> 7, n = e & 127;
        s_PH[i * LD128 + n] = gelu_f(s_PH[i * LD128 + n] + b1a[n]);
    }
    __syncthreads();

    // ---- MLP1b: h1(32x128) @ w1b(128x64) -> s_m1 raw ----
    mm_wave<32, 64, 128>(s_PH, LD128, w1b, 64, s_m1, LD64, wave);
    __syncthreads();
    for (int e = tid; e < 2048; e += 256) {
        int i = e >> 6, d = e & 63;
        s_m1[i * LD64 + d] = (s_m1[i * LD64 + d] + b1b[d]) * s_rmask[i];
    }
    __syncthreads();

    // ---- P = x_mlp1 @ w2a ; Q = xn @ w2a ----
    mm_wave<32, 128, 64>(s_m1, LD64, w2a, 128, s_PH, LD128, wave);
    mm_wave<32, 128, 64>(s_xn, LD64, w2a, 128, s_Q,  LD128, wave);
    __syncthreads();

    // u[k] = w2b[k,:] @ w3 ; P += b2a
    if (tid < 128)
        s_red[tid] = w2b[tid * 2 + 0] * w3[0] + w2b[tid * 2 + 1] * w3[1];
    for (int e = tid; e < 4096; e += 256) {
        int i = e >> 7, n = e & 127;
        s_PH[i * LD128 + n] += b2a[n];
    }
    __syncthreads();

    // ---- t[i,j] = sum_k gelu(P[i,k]-Q[j,k]) * u[k] + c0 ; store transposed ----
    // float4 LDS reads (rows are 16B aligned with LD128=132)
    const float c0 = b2b[0] * w3[0] + b2b[1] * w3[1];
    const float4* U = reinterpret_cast<const float4*>(s_red);
    for (int p = tid; p < 1024; p += 256) {
        int i = p >> 5, j = p & 31;
        const float4* Pi = reinterpret_cast<const float4*>(s_PH + i * LD128);
        const float4* Qj = reinterpret_cast<const float4*>(s_Q  + j * LD128);
        float acc = 0.f;
#pragma unroll 8
        for (int k4 = 0; k4 < 32; ++k4) {
            float4 pv = Pi[k4], qv = Qj[k4], uv = U[k4];
            acc += gelu_f(pv.x - qv.x) * uv.x;
            acc += gelu_f(pv.y - qv.y) * uv.y;
            acc += gelu_f(pv.z - qv.z) * uv.z;
            acc += gelu_f(pv.w - qv.w) * uv.w;
        }
        s_tT[j * LD32 + i] = acc + c0;
    }
    __syncthreads();

    // ---- proj = tT(32x32) @ x_mlp1(32x64) -> reuse s_xn as raw proj ----
    mm_wave<32, 64, 32>(s_tT, LD32, s_m1, LD64, s_xn, LD64, wave);
    __syncthreads();

    // ---- out = (proj + b3 + x_orig) * mask ----
    const float bb3 = b3[0];
    float* outp = out + (size_t)set * 2048;
    for (int e = tid; e < 2048; e += 256) {
        int i = e >> 6, d = e & 63;
        outp[e] = (s_xn[i * LD64 + d] + bb3 + s_x[i * LD64 + d]) * s_rmask[i];
    }
}

extern "C" void kernel_launch(void* const* d_in, const int* in_sizes, int n_in,
                              void* d_out, int out_size, void* d_ws, size_t ws_size,
                              hipStream_t stream) {
    const float* x   = (const float*)d_in[0];
    const float* xsz = (const float*)d_in[1];
    const float* w1a = (const float*)d_in[2];
    const float* b1a = (const float*)d_in[3];
    const float* w1b = (const float*)d_in[4];
    const float* b1b = (const float*)d_in[5];
    const float* w2a = (const float*)d_in[6];
    const float* b2a = (const float*)d_in[7];
    const float* w2b = (const float*)d_in[8];
    const float* b2b = (const float*)d_in[9];
    const float* w3  = (const float*)d_in[10];
    const float* b3  = (const float*)d_in[11];
    float* out = (float*)d_out;

    smn_fused<<<dim3(256), dim3(256), 0, stream>>>(
        x, xsz, w1a, b1a, w1b, b1b, w2a, b2a, w2b, b2b, w3, b3, out);
}
